// GRUFFNBlock_2138893714104
// MI455X (gfx1250) — matrix operand order repaired and checked by reading, not compiled, not run on hardware
//
#include <hip/hip_runtime.h>
#include <hip/hip_bf16.h>

// Problem dimensions (fixed by the reference).
#define B_  32
#define S_  512
#define D_  512
#define H_  2048
#define G3_ (3 * H_)        // 6144
#define M_  (B_ * S_)       // 16384

typedef __bf16 bf16;
typedef __attribute__((ext_vector_type(16))) __bf16 v16bf;
typedef __attribute__((ext_vector_type(8)))  __bf16 v8bf;
typedef __attribute__((ext_vector_type(8)))  float  v8f;

union Frag { v16bf v; v8bf h[2]; };

// ---------------------------------------------------------------------------
// WMMA fragment loaders (wave32; A and B share one contiguous k order).
// A 16x32 (MxK): lane l -> m = l&15, half = l>>4; holds K = half*16+0..15
//   -> contiguous 32B chunk per lane (matches B).
// B 32x16 (KxN) row-major W (N x K): lane l -> n = l&15, half = l>>4;
//   holds W[n][k0 + half*16 .. +15] -> contiguous 32B chunk per lane.
// C/D 16x16 f32: element (i, lane) -> M = i + 8*(lane>>4), N = lane&15.
// ---------------------------------------------------------------------------
__device__ __forceinline__ v16bf load_a_frag(const bf16* __restrict__ rowp, int k0) {
    Frag f;
    f.h[0] = *(const v8bf*)(rowp + k0);
    f.h[1] = *(const v8bf*)(rowp + k0 + 8);
    return f.v;
}

__device__ __forceinline__ v16bf load_b_frag(const bf16* __restrict__ rowp, int k0) {
    Frag f;
    f.h[0] = *(const v8bf*)(rowp + k0);
    f.h[1] = *(const v8bf*)(rowp + k0 + 8);
    return f.v;
}

// Packed B fragment: 16 contiguous bf16 (32B) per lane.
__device__ __forceinline__ v16bf load_b_packed(const bf16* __restrict__ p) {
    Frag f;
    f.h[0] = *(const v8bf*)(p);
    f.h[1] = *(const v8bf*)(p + 8);
    return f.v;
}

#define WMMA_BF16(A, Bm, C) \
    __builtin_amdgcn_wmma_f32_16x16x32_bf16(false, (A), false, (Bm), (short)0, (C), false, false)

// ---------------------------------------------------------------------------
// f32 -> bf16 conversion (grid-stride)
// ---------------------------------------------------------------------------
__global__ void cvt_f32_bf16(const float* __restrict__ src, bf16* __restrict__ dst, int n) {
    int i = blockIdx.x * blockDim.x + threadIdx.x;
    int stride = gridDim.x * blockDim.x;
    for (; i < n; i += stride) dst[i] = (bf16)src[i];
}

// Pack W_hh (6144 x 2048 f32, row-major N x K) into fragment-sequential bf16:
//   dst[((jt*3 + gate)*64 + kstep)*512 + lane*16 + koff]
// so a wave's B fragment for (jtile, gate, kstep) is one contiguous 1 KB blob
// (lane l owns bytes [l*32, l*32+32)) -> perfectly coalesced L2 streaming.
__global__ void pack_whh(const float* __restrict__ src, bf16* __restrict__ dst) {
    int idx = blockIdx.x * blockDim.x + threadIdx.x;
    int stride = gridDim.x * blockDim.x;
    const int total = G3_ * H_;
    for (; idx < total; idx += stride) {
        int n = idx / H_, k = idx - n * H_;
        int gate  = n >> 11;           // n / 2048
        int jt    = (n & 2047) >> 4;   // (n % 2048) / 16
        int nl    = n & 15;
        int kstep = k >> 5;            // k / 32
        int khalf = (k >> 4) & 1;
        int koff  = k & 15;
        int lane  = khalf * 16 + nl;
        size_t d = ((size_t)((jt * 3 + gate) * 64 + kstep)) * 512 + lane * 16 + koff;
        dst[d] = (bf16)src[idx];
    }
}

__global__ void init_barrier(unsigned* bar) {
    if (threadIdx.x < 2) bar[threadIdx.x] = 0u;
}

// ---------------------------------------------------------------------------
// Phase 1: x_gates(bf16)[m, g] = x[m,:] . W_ih[g,:] + b_ih[g]
//   M=16384, N=6144, K=512.  Wave tile 16x64, double-buffered fragments.
// ---------------------------------------------------------------------------
__global__ __launch_bounds__(256) void gemm_xgates(
    const bf16* __restrict__ xb, const bf16* __restrict__ Wih,
    const float* __restrict__ bih, bf16* __restrict__ xg) {
    const int lane = threadIdx.x & 31;
    const int wave = blockIdx.x * (blockDim.x >> 5) + (threadIdx.x >> 5);
    const int NG = G3_ / 64;                       // 96 N-groups
    const int mt = wave / NG;
    const int n0 = (wave % NG) * 64;

    const bf16* arow = xb + (size_t)(mt * 16 + (lane & 15)) * D_ + ((lane >> 4) * 16);
    const bf16* brow[4];
#pragma unroll
    for (int j = 0; j < 4; ++j)
        brow[j] = Wih + (size_t)(n0 + j * 16 + (lane & 15)) * D_ + ((lane >> 4) * 16);

    v8f acc[4] = {};
    v16bf a0 = load_a_frag(arow, 0), a1;
    v16bf b0[4], b1[4];
#pragma unroll
    for (int j = 0; j < 4; ++j) b0[j] = load_b_frag(brow[j], 0);

    for (int k0 = 0; k0 < D_; k0 += 64) {
        a1 = load_a_frag(arow, k0 + 32);
#pragma unroll
        for (int j = 0; j < 4; ++j) b1[j] = load_b_frag(brow[j], k0 + 32);
#pragma unroll
        for (int j = 0; j < 4; ++j) acc[j] = WMMA_BF16(a0, b0[j], acc[j]);
        const int kn = (k0 + 64 < D_) ? (k0 + 64) : 0;   // last iter: harmless reload
        a0 = load_a_frag(arow, kn);
#pragma unroll
        for (int j = 0; j < 4; ++j) b0[j] = load_b_frag(brow[j], kn);
#pragma unroll
        for (int j = 0; j < 4; ++j) acc[j] = WMMA_BF16(a1, b1[j], acc[j]);
    }

    const int mrow = mt * 16 + 8 * (lane >> 4);
#pragma unroll
    for (int j = 0; j < 4; ++j) {
        const int col = n0 + j * 16 + (lane & 15);
        const float bias = bih[col];
#pragma unroll
        for (int i = 0; i < 8; ++i)
            xg[(size_t)(mrow + i) * G3_ + col] = (bf16)(acc[j][i] + bias);
    }
}

// ---------------------------------------------------------------------------
// Device-scope grid barrier (persistent kernel, 16 co-resident blocks).
// ---------------------------------------------------------------------------
__device__ __forceinline__ void grid_sync(unsigned* bar, unsigned nblk) {
    __syncthreads();
    if (threadIdx.x == 0) {
        __builtin_amdgcn_fence(__ATOMIC_RELEASE, "agent");
        unsigned gen  = __hip_atomic_load(&bar[1], __ATOMIC_RELAXED, __HIP_MEMORY_SCOPE_AGENT);
        unsigned prev = __hip_atomic_fetch_add(&bar[0], 1u, __ATOMIC_RELAXED, __HIP_MEMORY_SCOPE_AGENT);
        if (prev == nblk - 1u) {
            __hip_atomic_store(&bar[0], 0u, __ATOMIC_RELAXED, __HIP_MEMORY_SCOPE_AGENT);
            __hip_atomic_store(&bar[1], gen + 1u, __ATOMIC_RELEASE, __HIP_MEMORY_SCOPE_AGENT);
        } else {
            while (__hip_atomic_load(&bar[1], __ATOMIC_ACQUIRE, __HIP_MEMORY_SCOPE_AGENT) == gen)
                __builtin_amdgcn_s_sleep(2);
        }
        __builtin_amdgcn_fence(__ATOMIC_ACQUIRE, "agent");
    }
    __syncthreads();
}

// ---------------------------------------------------------------------------
// Phase 2: persistent GRU scan. 16 blocks x 256 thr = 128 waves.
// Each wave owns the full 32x16 tile (all batches, one j-tile): 6 f32
// accumulators (r,z,n x 2 batch-halves) sharing 3 packed B fragments ->
// 6 WMMAs per k-step with half the B-side L2 traffic of the 16-row scheme.
// The whole 32x2048 bf16 h panel is async-copied to LDS per step (128 KB,
// ASYNCcnt-tracked) and shared by the block's 8 waves; A frags come from
// ds_load_b128.  x_gates lines are prefetched (global_prefetch_b8) before
// the K-loop so HBM latency hides under the GEMM.  One grid barrier per
// step with ping-pong bf16 h buffers; h state and gate math stay in f32
// registers.
// ---------------------------------------------------------------------------
__global__ __launch_bounds__(256) void gru_scan(
    const bf16* __restrict__ xg, const bf16* __restrict__ Whh_p,
    const float* __restrict__ bhh, bf16* __restrict__ hbuf /* 2 x B_*H_ */,
    bf16* __restrict__ relu_out, unsigned* __restrict__ bar) {
    extern __shared__ __align__(16) char smem[];
    bf16* panel = (bf16*)smem;                               // 32 x 2048 bf16 = 128 KB

    const int lane = threadIdx.x & 31;
    const int jt   = blockIdx.x * 8 + (threadIdx.x >> 5);    // 0..127 (j-tile)
    const int j0   = jt * 16;
    const int col  = j0 + (lane & 15);
    const int rsub = 8 * (lane >> 4);                        // row-in-16 base for C tiles

    // Zero h0 ping buffer (collectively covers all 32 x 2048).
#pragma unroll
    for (int half = 0; half < 2; ++half)
#pragma unroll
        for (int i = 0; i < 8; ++i)
            hbuf[(size_t)(half * 16 + rsub + i) * H_ + col] = (bf16)0.f;

    const float bhr = bhh[col];
    const float bhz = bhh[H_ + col];
    const float bhn = bhh[2 * H_ + col];

    // Packed W_hh fragment streams (r,z,n): 512 contiguous elements per
    // kstep, lane owns [lane*16, lane*16+16).
    const bf16* Wr = Whh_p + ((size_t)(jt * 3 + 0) * 64) * 512 + lane * 16;
    const bf16* Wz = Whh_p + ((size_t)(jt * 3 + 1) * 64) * 512 + lane * 16;
    const bf16* Wn = Whh_p + ((size_t)(jt * 3 + 2) * 64) * 512 + lane * 16;

    // LDS A row pointers: batch-half 0 = panel rows 0..15, half 1 = 16..31.
    const bf16* arow0 = panel + (size_t)(lane & 15) * H_ + ((lane >> 4) * 16);
    const bf16* arow1 = arow0 + (size_t)16 * H_;

    v8f h0 = {}, h1 = {};   // f32 master h state (two 16x16 tiles)

    grid_sync(bar, gridDim.x);

    for (int t = 0; t < S_; ++t) {
        const bf16* hb  = hbuf + (size_t)(t & 1) * (B_ * H_);
        bf16*       hbn = hbuf + (size_t)((t + 1) & 1) * (B_ * H_);

        // ---- Stage the full 32 x 2048 h panel into LDS via the CDNA5
        //      async-to-LDS engine (tracked by ASYNCcnt).
        {
            unsigned lbase = (unsigned)(uintptr_t)panel;     // LDS offset = addr[31:0]
#pragma unroll
            for (int r = 0; r < 32; ++r) {
                int idx = (threadIdx.x + 256 * r) * 8;       // element index (8 bf16 = 16B)
                unsigned long long ga = (unsigned long long)(uintptr_t)(hb + idx);
                unsigned la = lbase + idx * 2;
                asm volatile("global_load_async_to_lds_b128 %0, %1, off"
                             :: "v"(la), "v"(ga) : "memory");
            }
        }

        // ---- Prefetch this step's x_gates lines (HBM) under the GEMM.
#pragma unroll
        for (int half = 0; half < 2; ++half) {
            const bf16* p = xg + ((size_t)(half * 16 + rsub) * S_ + t) * G3_ + col;
#pragma unroll
            for (int i = 0; i < 8; ++i) {
                const bf16* q = p + (size_t)i * S_ * G3_;
                __builtin_prefetch(q, 0, 1);
                __builtin_prefetch(q + H_, 0, 1);
                __builtin_prefetch(q + 2 * H_, 0, 1);
            }
        }

        asm volatile("s_wait_asynccnt 0" ::: "memory");
        __syncthreads();

        // ---- Double-buffered K loop: 64 ksteps of 32, 6 WMMAs each.
        v8f ar0 = {}, az0 = {}, an0 = {}, ar1 = {}, az1 = {}, an1 = {};
        v16bf brA = load_b_packed(Wr), bzA = load_b_packed(Wz), bnA = load_b_packed(Wn);
        v16bf brB, bzB, bnB;
        for (int ks = 0; ks < 64; ks += 2) {
            brB = load_b_packed(Wr + (ks + 1) * 512);
            bzB = load_b_packed(Wz + (ks + 1) * 512);
            bnB = load_b_packed(Wn + (ks + 1) * 512);
            {
                v16bf a0 = load_a_frag(arow0, ks * 32);
                v16bf a1 = load_a_frag(arow1, ks * 32);
                ar0 = WMMA_BF16(a0, brA, ar0);  ar1 = WMMA_BF16(a1, brA, ar1);
                az0 = WMMA_BF16(a0, bzA, az0);  az1 = WMMA_BF16(a1, bzA, az1);
                an0 = WMMA_BF16(a0, bnA, an0);  an1 = WMMA_BF16(a1, bnA, an1);
            }
            const int kn = (ks + 2 < 64) ? (ks + 2) : 0;     // last iter: harmless reload
            brA = load_b_packed(Wr + kn * 512);
            bzA = load_b_packed(Wz + kn * 512);
            bnA = load_b_packed(Wn + kn * 512);
            {
                v16bf a0 = load_a_frag(arow0, (ks + 1) * 32);
                v16bf a1 = load_a_frag(arow1, (ks + 1) * 32);
                ar0 = WMMA_BF16(a0, brB, ar0);  ar1 = WMMA_BF16(a1, brB, ar1);
                az0 = WMMA_BF16(a0, bzB, az0);  az1 = WMMA_BF16(a1, bzB, az1);
                an0 = WMMA_BF16(a0, bnB, an0);  an1 = WMMA_BF16(a1, bnB, an1);
            }
        }

        // ---- Gate math in f32 for both batch halves; x_gates row m = b*S + t.
#pragma unroll
        for (int half = 0; half < 2; ++half) {
            const v8f& ar = half ? ar1 : ar0;
            const v8f& az = half ? az1 : az0;
            const v8f& an = half ? an1 : an0;
            v8f&       h  = half ? h1  : h0;
            const int brow = half * 16 + rsub;
            const bf16* xgp = xg + ((size_t)brow * S_ + t) * G3_ + col;
#pragma unroll
            for (int i = 0; i < 8; ++i) {
                const size_t off = (size_t)i * S_ * G3_;
                const float xr = (float)xgp[off];
                const float xz = (float)xgp[off + H_];
                const float xn = (float)xgp[off + 2 * H_];
                const float r  = 1.f / (1.f + __expf(-(xr + ar[i] + bhr)));
                const float z  = 1.f / (1.f + __expf(-(xz + az[i] + bhz)));
                const float n  = tanhf(xn + r * (an[i] + bhn));
                const float hv = (1.f - z) * n + z * h[i];
                h[i] = hv;
                hbn[(size_t)(brow + i) * H_ + col] = (bf16)hv;
                relu_out[((size_t)(brow + i) * S_ + t) * H_ + col] = (bf16)fmaxf(hv, 0.f);
            }
        }
        grid_sync(bar, gridDim.x);
    }
}

// ---------------------------------------------------------------------------
// Phase 3: out(f32)[m, d] = relu_h[m,:] . W_proj[d,:] + b_proj[d]
//   M=16384, N=512, K=2048.  Wave tile 16x64, double-buffered fragments.
// ---------------------------------------------------------------------------
__global__ __launch_bounds__(256) void gemm_proj(
    const bf16* __restrict__ ab, const bf16* __restrict__ Wp,
    const float* __restrict__ bp, float* __restrict__ out) {
    const int lane = threadIdx.x & 31;
    const int wave = blockIdx.x * (blockDim.x >> 5) + (threadIdx.x >> 5);
    const int NG = D_ / 64;                        // 8 N-groups
    const int mt = wave / NG;
    const int n0 = (wave % NG) * 64;

    const bf16* arow = ab + (size_t)(mt * 16 + (lane & 15)) * H_ + ((lane >> 4) * 16);
    const bf16* brow[4];
#pragma unroll
    for (int j = 0; j < 4; ++j)
        brow[j] = Wp + (size_t)(n0 + j * 16 + (lane & 15)) * H_ + ((lane >> 4) * 16);

    v8f acc[4] = {};
    v16bf a0 = load_a_frag(arow, 0), a1;
    v16bf b0[4], b1[4];
#pragma unroll
    for (int j = 0; j < 4; ++j) b0[j] = load_b_frag(brow[j], 0);

    for (int k0 = 0; k0 < H_; k0 += 64) {
        a1 = load_a_frag(arow, k0 + 32);
#pragma unroll
        for (int j = 0; j < 4; ++j) b1[j] = load_b_frag(brow[j], k0 + 32);
#pragma unroll
        for (int j = 0; j < 4; ++j) acc[j] = WMMA_BF16(a0, b0[j], acc[j]);
        const int kn = (k0 + 64 < H_) ? (k0 + 64) : 0;
        a0 = load_a_frag(arow, kn);
#pragma unroll
        for (int j = 0; j < 4; ++j) b0[j] = load_b_frag(brow[j], kn);
#pragma unroll
        for (int j = 0; j < 4; ++j) acc[j] = WMMA_BF16(a1, b1[j], acc[j]);
    }

    const int mrow = mt * 16 + 8 * (lane >> 4);
#pragma unroll
    for (int j = 0; j < 4; ++j) {
        const int col = n0 + j * 16 + (lane & 15);
        const float bias = bp[col];
#pragma unroll
        for (int i = 0; i < 8; ++i)
            out[(size_t)(mrow + i) * D_ + col] = acc[j][i] + bias;
    }
}

// ---------------------------------------------------------------------------
extern "C" void kernel_launch(void* const* d_in, const int* in_sizes, int n_in,
                              void* d_out, int out_size, void* d_ws, size_t ws_size,
                              hipStream_t stream) {
    const float* x      = (const float*)d_in[0];
    const float* W_ih   = (const float*)d_in[1];
    const float* W_hh   = (const float*)d_in[2];
    const float* b_ih   = (const float*)d_in[3];
    const float* b_hh   = (const float*)d_in[4];
    const float* W_proj = (const float*)d_in[5];
    const float* b_proj = (const float*)d_in[6];
    float* out = (float*)d_out;

    // Workspace carve-up (all buffers written before read each call).
    char* ws = (char*)d_ws;
    size_t off = 0;
    auto carve = [&](size_t bytes) {
        void* p = ws + off;
        off += (bytes + 255) & ~(size_t)255;
        return p;
    };
    bf16* xb      = (bf16*)carve(sizeof(bf16) * (size_t)M_ * D_);     //  16 MB
    bf16* Wih_b   = (bf16*)carve(sizeof(bf16) * (size_t)G3_ * D_);    // 6.3 MB
    bf16* Whh_p   = (bf16*)carve(sizeof(bf16) * (size_t)G3_ * H_);    //  25 MB (packed)
    bf16* Wp_b    = (bf16*)carve(sizeof(bf16) * (size_t)D_ * H_);     //   2 MB
    bf16* xg      = (bf16*)carve(sizeof(bf16) * (size_t)M_ * G3_);    // 201 MB
    bf16* relu_b  = (bf16*)carve(sizeof(bf16) * (size_t)M_ * H_);     //  64 MB
    bf16* hbuf    = (bf16*)carve(sizeof(bf16) * 2 * (size_t)B_ * H_); // 256 KB
    unsigned* bar = (unsigned*)carve(256);

    // 1) Convert / pack weights and activations to bf16.
    cvt_f32_bf16<<<4096, 256, 0, stream>>>(x,      xb,    M_ * D_);
    cvt_f32_bf16<<<4096, 256, 0, stream>>>(W_ih,   Wih_b, G3_ * D_);
    pack_whh    <<<8192, 256, 0, stream>>>(W_hh,   Whh_p);
    cvt_f32_bf16<<<2048, 256, 0, stream>>>(W_proj, Wp_b,  D_ * H_);
    init_barrier<<<1, 32, 0, stream>>>(bar);

    // 2) Input-side gate GEMM: (16384x512)x(512x6144) -> bf16 x_gates.
    gemm_xgates<<<12288, 256, 0, stream>>>(xb, Wih_b, b_ih, xg);

    // 3) Persistent GRU recurrence: 16 blocks x 256 thr (128 waves),
    //    128 KB dynamic LDS h panel per block.
    gru_scan<<<16, 256, 32 * H_ * sizeof(bf16), stream>>>(xg, Whh_p, b_hh, hbuf, relu_b, bar);

    // 4) Output projection: (16384x2048)x(2048x512) -> f32 out (+bias).
    gemm_proj<<<1024, 256, 0, stream>>>(relu_b, Wp_b, b_proj, out);
}
